// BasicQuantumAttention_33603824124333
// MI455X (gfx1250) — compile-verified, matrix operand order checked by reading, not run on hardware
//
#include <hip/hip_runtime.h>

// ---------------------------------------------------------------------------
// BasicQuantumAttention on MI455X (gfx1250): bf16 WMMA everywhere.
//   GEMM1: x[8192,3072] @ Wqkv^T -> qkv (bf16) + transposed V copy (vT)
//   Attn : block-sparse flash attention, 64-row blocks, V read from vT
//   GEMM2: attn[8192,1024] @ Wout^T -> out (f32, split real/imag)
// GEMMs: 128x256 block tile, 8 waves (2x4), 64x64 wave tile, K-step 32,
//        double-buffered LDS staging via GLOBAL_LOAD_ASYNC_TO_LDS_B128.
// ---------------------------------------------------------------------------

typedef __attribute__((ext_vector_type(16))) __bf16 v16bf;
typedef __attribute__((ext_vector_type(8)))  __bf16 v8bf;
typedef __attribute__((ext_vector_type(8)))  float  v8f;
typedef int v4i __attribute__((__vector_size__(4 * sizeof(int))));

#define B_   4
#define L_   2048
#define D_   512
#define M_   (B_ * L_)     /* 8192 rows */
#define K6_  (6 * D_)      /* 3072      */
#define D2_  (2 * D_)      /* 1024      */

// ---- async VMEM->LDS path (gfx1250), with graceful fallback ---------------
#if __has_builtin(__builtin_amdgcn_global_load_async_to_lds_b128)
#define HAVE_ASYNC_LDS 1
#else
#define HAVE_ASYNC_LDS 0
#endif

__device__ __forceinline__ void async_cp16(const void* g, void* l) {
#if HAVE_ASYNC_LDS
  __builtin_amdgcn_global_load_async_to_lds_b128((v4i*)g, (v4i*)l, 0, 0);
#else
  *(v8bf*)l = *(const v8bf*)g;
#endif
}

#if HAVE_ASYNC_LDS
#if __has_builtin(__builtin_amdgcn_s_wait_asynccnt)
#define WAIT_ASYNC(n) __builtin_amdgcn_s_wait_asynccnt(n)
#else
#define WAIT_ASYNC(n) asm volatile("s_wait_asynccnt %0" ::"n"(n) : "memory")
#endif
#else
#define WAIT_ASYNC(n) ((void)0)
#endif

union Frag16 { v16bf v; v8bf h[2]; };

// A/B fragment loader for V_WMMA_F32_16X16X32_BF16 from a K-contiguous
// row-major matrix: lane holds K chunks [k0+8*laneHi .. +7] and [+16 .. +23].
__device__ __forceinline__ v16bf load_frag(const __bf16* __restrict__ base,
                                           long rowBase, long ld, long k0,
                                           int lane15, int laneHi) {
  Frag16 f;
  const __bf16* p = base + (rowBase + lane15) * ld + k0 + (long)laneHi * 8;
  f.h[0] = *reinterpret_cast<const v8bf*>(p);
  f.h[1] = *reinterpret_cast<const v8bf*>(p + 16);
  return f.v;
}

__device__ __forceinline__ v8f wmma_bf16(v16bf a, v16bf b, v8f c) {
  return __builtin_amdgcn_wmma_f32_16x16x32_bf16(false, a, false, b,
                                                 (short)0, c, false, false);
}

__device__ __forceinline__ v8f vzero8() {
  v8f z = {0.f, 0.f, 0.f, 0.f, 0.f, 0.f, 0.f, 0.f};
  return z;
}

// ---------------------------------------------------------------------------
// Kernel 1: gather the 6 input streams into x_bf[8192,3072] (bf16)
// ---------------------------------------------------------------------------
__global__ __launch_bounds__(256) void pack_x_kernel(
    const float* __restrict__ qr, const float* __restrict__ qi,
    const float* __restrict__ kr, const float* __restrict__ ki,
    const float* __restrict__ vr, const float* __restrict__ vi,
    __bf16* __restrict__ x_bf) {
  long idx = (long)blockIdx.x * 256 + threadIdx.x;
  if (idx >= (long)M_ * K6_) return;
  long m = idx / K6_;
  int  c = (int)(idx % K6_);
  int  t = c >> 9, cc = c & 511;
  long off = m * D_ + cc;
  float v;
  switch (t) {
    case 0: v = qr[off]; break;
    case 1: v = qi[off]; break;
    case 2: v = kr[off]; break;
    case 3: v = ki[off]; break;
    case 4: v = vr[off]; break;
    default: v = vi[off]; break;
  }
  x_bf[idx] = (__bf16)v;
}

__global__ __launch_bounds__(256) void f32_to_bf16_kernel(
    const float* __restrict__ in, __bf16* __restrict__ out, long n) {
  long idx = (long)blockIdx.x * 256 + threadIdx.x;
  if (idx < n) out[idx] = (__bf16)in[idx];
}

// ---------------------------------------------------------------------------
// Shared GEMM core: acc[4][4] += A[128 rows] x W^T[256 rows], both row-major
// K-contiguous bf16.  LDS double buffer, async staging, 16 WMMA per K-step.
// ---------------------------------------------------------------------------
#define LDS_STRIDE 40                         /* bf16 elems, 80B: pad vs banks */
#define SA_ELEMS   (128 * LDS_STRIDE)         /* per buffer */
#define SB_ELEMS   (256 * LDS_STRIDE)
#define GEMM_SMEM  ((2 * SA_ELEMS + 2 * SB_ELEMS) * 2)   /* 61440 bytes */

__device__ __forceinline__ void gemm_core(
    const __bf16* __restrict__ A, const __bf16* __restrict__ W, long K,
    char* smem, long mBlock, long nBlock, int wm, int wn, int lane15,
    int laneHi, v8f acc[4][4]) {
  __bf16* sA = reinterpret_cast<__bf16*>(smem);
  __bf16* sB = sA + 2 * SA_ELEMS;
  const int tid = threadIdx.x;

  auto stage = [&](int buf, long kk) {
    __bf16* dA = sA + buf * SA_ELEMS;
    __bf16* dB = sB + buf * SB_ELEMS;
#pragma unroll
    for (int it = 0; it < 2; ++it) {          // A: 128 rows x 32 = 512 chunks
      int c = it * 256 + tid;
      int row = c >> 2, ko = (c & 3) * 8;
      async_cp16(A + (mBlock + row) * K + kk + ko, dA + row * LDS_STRIDE + ko);
    }
#pragma unroll
    for (int it = 0; it < 4; ++it) {          // B: 256 rows x 32 = 1024 chunks
      int c = it * 256 + tid;
      int row = c >> 2, ko = (c & 3) * 8;
      async_cp16(W + (nBlock + row) * K + kk + ko, dB + row * LDS_STRIDE + ko);
    }
  };

  stage(0, 0);                                // 6 async ops in flight / thread
  for (long kk = 0; kk < K; kk += 32) {
    const int buf = (int)((kk >> 5) & 1);
    const bool hasNext = (kk + 32) < K;
    if (hasNext) stage(buf ^ 1, kk + 32);     // overlap DMA with compute
    if (hasNext) { WAIT_ASYNC(6); } else { WAIT_ASYNC(0); }
    __syncthreads();

    v16bf af[4], bfr[4];
    const __bf16* pA = sA + buf * SA_ELEMS;
    const __bf16* pB = sB + buf * SB_ELEMS;
#pragma unroll
    for (int rt = 0; rt < 4; ++rt)
      af[rt] = load_frag(pA, wm * 64 + rt * 16, LDS_STRIDE, 0, lane15, laneHi);
#pragma unroll
    for (int ct = 0; ct < 4; ++ct)
      bfr[ct] = load_frag(pB, wn * 64 + ct * 16, LDS_STRIDE, 0, lane15, laneHi);
#pragma unroll
    for (int rt = 0; rt < 4; ++rt)
#pragma unroll
      for (int ct = 0; ct < 4; ++ct)
        acc[rt][ct] = wmma_bf16(af[rt], bfr[ct], acc[rt][ct]);
    __syncthreads();                          // release buf for re-staging
  }
}

// ---------------------------------------------------------------------------
// Kernel 2: qkv = x @ Wqkv^T + b (bf16 out).  For the V part (cols 2048..3071)
// also emit a transposed copy vT[dcol][m] (row stride M_=8192): one D-fragment
// lane holds 8 consecutive output rows at fixed col -> single v8bf store.
// ---------------------------------------------------------------------------
__global__ __launch_bounds__(256) void gemm_qkv_kernel(
    const __bf16* __restrict__ A, const __bf16* __restrict__ W,
    const float* __restrict__ bias, __bf16* __restrict__ C,
    __bf16* __restrict__ vT) {
  extern __shared__ __align__(16) char smem[];
  const int lane = threadIdx.x & 31, lane15 = lane & 15, laneHi = lane >> 4;
  const int wave = threadIdx.x >> 5;
  const int wm = wave >> 2, wn = wave & 3;
  const long mBlock = (long)blockIdx.y * 128;
  const long nBlock = (long)blockIdx.x * 256;

  v8f acc[4][4];
#pragma unroll
  for (int rt = 0; rt < 4; ++rt)
#pragma unroll
    for (int ct = 0; ct < 4; ++ct) acc[rt][ct] = vzero8();

  gemm_core(A, W, K6_, smem, mBlock, nBlock, wm, wn, lane15, laneHi, acc);

#pragma unroll
  for (int rt = 0; rt < 4; ++rt)
#pragma unroll
    for (int ct = 0; ct < 4; ++ct) {
      long col = nBlock + wn * 64 + ct * 16 + lane15;
      long m0  = mBlock + wm * 64 + rt * 16 + 8 * laneHi;   // 8 consecutive rows
      float bv = bias[col];
      Frag16 pack;
#pragma unroll
      for (int e = 0; e < 8; ++e) {
        float v = acc[rt][ct][e] + bv;
        C[(m0 + e) * K6_ + col] = (__bf16)v;
        pack.h[0][e] = (__bf16)v;
      }
      if (col >= 2 * D2_)                       // V columns -> transposed copy
        *reinterpret_cast<v8bf*>(vT + (col - 2 * D2_) * (long)M_ + m0) =
            pack.h[0];
    }
}

// ---------------------------------------------------------------------------
// Kernel 3: block-sparse flash attention for one (batch, stream, 64-row block)
// qkv layout per row: [q(0:512) qI(512:1024) k(1024:1536) kI(1536:2048) v vI]
// The mask is block-constant 64x64 (tril|eye), so one byte gates a block.
// V is consumed pre-transposed from vT[dcol][m] -> direct B-fragment loads.
// ---------------------------------------------------------------------------
#define S_STRIDE  68            /* floats  */
#define P_STRIDE  72            /* bf16    */
#define SS_BYTES  (64 * S_STRIDE * 4)          /* 17408 */
#define SP_BYTES  (64 * P_STRIDE * 2)          /* 9216  */
#define ATTN_SMEM (SS_BYTES + SP_BYTES + 3 * 64 * 4)      /* 27392 */

__global__ __launch_bounds__(256) void attn_kernel(
    const __bf16* __restrict__ qkv, const __bf16* __restrict__ vT,
    const unsigned char* __restrict__ mask_r,
    const unsigned char* __restrict__ mask_i, __bf16* __restrict__ attn_out) {
  extern __shared__ __align__(16) char smem[];
  float*  sS     = reinterpret_cast<float*>(smem);
  __bf16* sP     = reinterpret_cast<__bf16*>(smem + SS_BYTES);
  float*  sM     = reinterpret_cast<float*>(smem + SS_BYTES + SP_BYTES);
  float*  sL     = sM + 64;
  float*  sAlpha = sL + 64;

  const int qb = blockIdx.x;           // 0..31 query block
  const int s  = blockIdx.y;           // 0 real, 1 imag
  const int b  = blockIdx.z;           // batch
  const unsigned char* mask = s ? mask_i : mask_r;
  const long qcol = (long)s * 512;
  const long kcol = 1024 + (long)s * 512;
  const float scale = 0.044194173824159216f;   // 512^-0.5

  const int lane = threadIdx.x & 31, lane15 = lane & 15, laneHi = lane >> 4;
  const int wave = threadIdx.x >> 5;
  const int ti  = wave >> 1;                   // S row tile (shared by pair)
  const int tj0 = (wave & 1) * 2;              // S col tiles tj0, tj0+1

  if (threadIdx.x < 64) { sM[threadIdx.x] = -1e30f; sL[threadIdx.x] = 0.f; }

  v8f acc[4][4];                       // O accumulator, wave's 64-wide d slice
#pragma unroll
  for (int rt = 0; rt < 4; ++rt)
#pragma unroll
    for (int ct = 0; ct < 4; ++ct) acc[rt][ct] = vzero8();
  __syncthreads();

  for (int kb = 0; kb <= qb; ++kb) {
    if (!mask[(size_t)(qb * 64) * L_ + kb * 64]) continue;   // uniform branch

    // ---- S = scale * Q @ K^T: two col tiles per wave, shared Q fragment ----
    {
      v8f sacc0 = vzero8(), sacc1 = vzero8();
      for (int dk = 0; dk < 512; dk += 32) {
        v16bf aq  = load_frag(qkv + qcol, (long)b * L_ + qb * 64 + ti * 16,
                              K6_, dk, lane15, laneHi);
        v16bf bk0 = load_frag(qkv + kcol, (long)b * L_ + kb * 64 + tj0 * 16,
                              K6_, dk, lane15, laneHi);
        v16bf bk1 = load_frag(qkv + kcol,
                              (long)b * L_ + kb * 64 + (tj0 + 1) * 16, K6_, dk,
                              lane15, laneHi);
        sacc0 = wmma_bf16(aq, bk0, sacc0);
        sacc1 = wmma_bf16(aq, bk1, sacc1);
      }
#pragma unroll
      for (int e = 0; e < 8; ++e) {
        int rr = (ti * 16 + e + 8 * laneHi) * S_STRIDE;
        sS[rr + tj0 * 16 + lane15]       = sacc0[e] * scale;
        sS[rr + (tj0 + 1) * 16 + lane15] = sacc1[e] * scale;
      }
    }
    __syncthreads();

    // ---- streaming softmax, one row per thread (rows 0..63) ----
    if (threadIdx.x < 64) {
      const int r = threadIdx.x;
      float mOld = sM[r], mNew = mOld;
      for (int j = 0; j < 64; ++j) mNew = fmaxf(mNew, sS[r * S_STRIDE + j]);
      float alpha = __expf(mOld - mNew);
      float lsum = 0.f;
      for (int j = 0; j < 64; ++j) {
        float e = __expf(sS[r * S_STRIDE + j] - mNew);
        lsum += e;
        sP[r * P_STRIDE + j] = (__bf16)e;
      }
      sM[r] = mNew;
      sL[r] = sL[r] * alpha + lsum;
      sAlpha[r] = alpha;
    }
    __syncthreads();

    // ---- rescale O by alpha, then O += P @ V (K=64 -> 2 WMMA k-steps) ----
#pragma unroll
    for (int rt = 0; rt < 4; ++rt) {
      float al[8];
#pragma unroll
      for (int e = 0; e < 8; ++e) al[e] = sAlpha[rt * 16 + e + 8 * laneHi];
#pragma unroll
      for (int ct = 0; ct < 4; ++ct)
#pragma unroll
        for (int e = 0; e < 8; ++e) acc[rt][ct][e] *= al[e];
    }
#pragma unroll
    for (int ks = 0; ks < 2; ++ks) {
      v16bf ap[4], bv[4];
#pragma unroll
      for (int rt = 0; rt < 4; ++rt)
        ap[rt] = load_frag(sP, rt * 16, P_STRIDE, ks * 32, lane15, laneHi);
#pragma unroll
      for (int ct = 0; ct < 4; ++ct)    // B-fragments straight from global vT
        bv[ct] = load_frag(vT, (long)s * 512 + wave * 64 + ct * 16, M_,
                           (long)b * L_ + kb * 64 + ks * 32, lane15, laneHi);
#pragma unroll
      for (int rt = 0; rt < 4; ++rt)
#pragma unroll
        for (int ct = 0; ct < 4; ++ct)
          acc[rt][ct] = wmma_bf16(ap[rt], bv[ct], acc[rt][ct]);
    }
    // no trailing barrier needed: the pre-P@V barrier orders this softmax's sS reads
    // vs next sS writes; next iteration's pre-softmax barrier orders sP/sAlpha.
  }
  __syncthreads();                      // sL stable before finalize

  // ---- finalize: O /= l, write bf16 into attn_bf[m, s*512 + dcol] ----
#pragma unroll
  for (int rt = 0; rt < 4; ++rt)
#pragma unroll
    for (int e = 0; e < 8; ++e) {
      int row = rt * 16 + e + 8 * laneHi;
      float inv = 1.0f / sL[row];     // diagonal block always active -> l > 0
#pragma unroll
      for (int ct = 0; ct < 4; ++ct) {
        long col = (long)s * 512 + wave * 64 + ct * 16 + lane15;
        attn_out[((long)b * L_ + qb * 64 + row) * D2_ + col] =
            (__bf16)(acc[rt][ct][e] * inv);
      }
    }
}

// ---------------------------------------------------------------------------
// Kernel 4: out = attn @ Wout^T + b, split into (real, imag) f32 outputs
// ---------------------------------------------------------------------------
__global__ __launch_bounds__(256) void gemm_out_kernel(
    const __bf16* __restrict__ A, const __bf16* __restrict__ W,
    const float* __restrict__ bias, float* __restrict__ out) {
  extern __shared__ __align__(16) char smem[];
  const int lane = threadIdx.x & 31, lane15 = lane & 15, laneHi = lane >> 4;
  const int wave = threadIdx.x >> 5;
  const int wm = wave >> 2, wn = wave & 3;
  const long mBlock = (long)blockIdx.y * 128;
  const long nBlock = (long)blockIdx.x * 256;

  v8f acc[4][4];
#pragma unroll
  for (int rt = 0; rt < 4; ++rt)
#pragma unroll
    for (int ct = 0; ct < 4; ++ct) acc[rt][ct] = vzero8();

  gemm_core(A, W, D2_, smem, mBlock, nBlock, wm, wn, lane15, laneHi, acc);

  float* out_imag = out + (long)M_ * D_;   // second tuple element, flat
#pragma unroll
  for (int rt = 0; rt < 4; ++rt)
#pragma unroll
    for (int ct = 0; ct < 4; ++ct) {
      long col = nBlock + wn * 64 + ct * 16 + lane15;
      float bv = bias[col];
#pragma unroll
      for (int e = 0; e < 8; ++e) {
        long row = mBlock + wm * 64 + rt * 16 + e + 8 * laneHi;
        float val = acc[rt][ct][e] + bv;
        if (col < D_) out[row * D_ + col] = val;
        else          out_imag[row * D_ + (col - D_)] = val;
      }
    }
}

// ---------------------------------------------------------------------------
extern "C" void kernel_launch(void* const* d_in, const int* in_sizes, int n_in,
                              void* d_out, int out_size, void* d_ws,
                              size_t ws_size, hipStream_t stream) {
  (void)in_sizes; (void)n_in; (void)out_size; (void)ws_size;
  const float* qr = (const float*)d_in[0];
  const float* qi = (const float*)d_in[1];
  const float* kr = (const float*)d_in[2];
  const float* ki = (const float*)d_in[3];
  const float* vr = (const float*)d_in[4];
  const float* vi = (const float*)d_in[5];
  const float* Wqkv = (const float*)d_in[6];
  const float* bqkv = (const float*)d_in[7];
  const float* Wout = (const float*)d_in[8];
  const float* bout = (const float*)d_in[9];
  const unsigned char* mr = (const unsigned char*)d_in[10];
  const unsigned char* mi = (const unsigned char*)d_in[11];

  // workspace (bytes): x_bf | Wqkv_bf | qkv_bf | attn_bf | Wout_bf | vT
  char* ws = (char*)d_ws;
  __bf16* x_bf    = (__bf16*)(ws);                         // 8192*3072*2
  __bf16* Wqkv_bf = (__bf16*)(ws + 50331648);              // 3072*3072*2
  __bf16* qkv_bf  = (__bf16*)(ws + 69206016);              // 8192*3072*2
  __bf16* attn_bf = (__bf16*)(ws + 119537664);             // 8192*1024*2
  __bf16* Wout_bf = (__bf16*)(ws + 136314880);             // 1024*1024*2
  __bf16* vT      = (__bf16*)(ws + 138412032);             // 1024*8192*2
  float* out = (float*)d_out;

  long nx = (long)M_ * K6_;
  pack_x_kernel<<<(unsigned)((nx + 255) / 256), 256, 0, stream>>>(
      qr, qi, kr, ki, vr, vi, x_bf);
  f32_to_bf16_kernel<<<(unsigned)(((long)K6_ * K6_ + 255) / 256), 256, 0,
                       stream>>>(Wqkv, Wqkv_bf, (long)K6_ * K6_);
  f32_to_bf16_kernel<<<(unsigned)(((long)D2_ * D2_ + 255) / 256), 256, 0,
                       stream>>>(Wout, Wout_bf, (long)D2_ * D2_);

  gemm_qkv_kernel<<<dim3(K6_ / 256, M_ / 128), 256, GEMM_SMEM, stream>>>(
      x_bf, Wqkv_bf, bqkv, qkv_bf, vT);

  attn_kernel<<<dim3(L_ / 64, 2, B_), 256, ATTN_SMEM, stream>>>(
      qkv_bf, vT, mr, mi, attn_bf);

  gemm_out_kernel<<<dim3(D2_ / 256, M_ / 128), 256, GEMM_SMEM, stream>>>(
      attn_bf, Wout_bf, bout, out);
}
